// TwoAreaCTRNN_58445914964037
// MI455X (gfx1250) — matrix operand order repaired and checked by reading, not compiled, not run on hardware
//
#include <hip/hip_runtime.h>
#include <hip/hip_bf16.h>

// Two-area CTRNN forward scan for MI455X (gfx1250).
// seq_lengths masking is h*m + stop_grad(h)*(1-m) == h -> forward no-op.
//
// v3: fix "illegal VGPR to SGPR copy" — the area-selected weight pointer is
// wave-uniform but threadIdx-derived, so it must be legalized with
// readfirstlane before it can be tied to an "s" inline-asm constraint.
//
// Design (unchanged from v2):
//  - W11/W22 staged in LDS (256KB) + 32KB h double-buffer = 288KB of the
//    320KB/WGP budget; W21*.5 / W12*.5 / Wi stream from L2 each step.
//  - anti-LICM asm so per-step weight loads stay in-loop (no spill reloads).
//  - row-rotation swizzle (k' = (k + row*8) & 255) on LDS tiles to avoid
//    16-way bank conflicts.
//  - 512 threads = 16 waves, 2 N-tiles per wave: each A fragment feeds two
//    WMMAs; 4 waves/SIMD doubles the VGPR budget so nothing spills.

typedef __attribute__((ext_vector_type(16))) __bf16 v16bf;
typedef __attribute__((ext_vector_type(8)))  __bf16 v8bf;
typedef __attribute__((ext_vector_type(8)))  float  v8f;

static constexpr int Bn = 64, Tn = 1024, In = 128, Hn = 256;
static constexpr float ADEC = 16.67f / 100.0f;   // A1 == A2

// ws layout in bf16 elements
static constexpr size_t OFF_WI  = 0;        // 256x128
static constexpr size_t OFF_W11 = 32768;    // 256x256
static constexpr size_t OFF_W22 = 98304;    // 256x256
static constexpr size_t OFF_W12 = 163840;   // 256x256, pre-scaled 0.5
static constexpr size_t OFF_W21 = 229376;   // 256x256, pre-scaled 0.5
static constexpr size_t OFF_X   = 294912;   // [64][1024][128]

__device__ __forceinline__ v16bf frag_cat(const __bf16* lo, const __bf16* hi) {
  v8bf a = *(const v8bf*)lo;
  v8bf b = *(const v8bf*)hi;
  v16bf r;
#pragma unroll
  for (int i = 0; i < 8; ++i) { r[i] = a[i]; r[i + 8] = b[i]; }
  return r;
}

// Assert wave-uniformity of a pointer so it lives in an SGPR pair (legal to
// tie to an "s" asm constraint; keeps saddr-form global loads).
__device__ __forceinline__ unsigned long long uniform_u64(unsigned long long p) {
  unsigned int lo = (unsigned int)__builtin_amdgcn_readfirstlane((int)(p & 0xffffffffull));
  unsigned int hi = (unsigned int)__builtin_amdgcn_readfirstlane((int)(p >> 32));
  return ((unsigned long long)hi << 32) | (unsigned long long)lo;
}

__global__ void cvt_bf16_kernel(const float* __restrict__ src,
                                unsigned short* __restrict__ dst,
                                int n, float scale) {
  int i = blockIdx.x * blockDim.x + threadIdx.x;
  if (i < n) {
    __hip_bfloat16 h = __float2bfloat16(src[i] * scale);
    dst[i] = __builtin_bit_cast(unsigned short, h);
  }
}

__global__ void __launch_bounds__(512)
ctrnn_scan_kernel(const unsigned short* __restrict__ ws,
                  const float* __restrict__ bi,
                  float* __restrict__ out) {
  const __bf16* Wi   = (const __bf16*)(ws + OFF_WI);
  const __bf16* W11g = (const __bf16*)(ws + OFF_W11);
  const __bf16* W22g = (const __bf16*)(ws + OFF_W22);
  const __bf16* W12  = (const __bf16*)(ws + OFF_W12);  // 0.5 folded in
  const __bf16* W21  = (const __bf16*)(ws + OFF_W21);  // 0.5 folded in
  const __bf16* Xb   = (const __bf16*)(ws + OFF_X);

  // row-rotated weight tiles: element (r,k) lives at [r][(k + r*8) & 255]
  __shared__ __align__(16) unsigned short sW[2][256][256];      // W11, W22 (256KB)
  // bf16 h state, double buffered, same row rotation (32KB)
  __shared__ __align__(16) unsigned short hbuf[2][2][16][256];

  const int tid  = threadIdx.x;
  const int lane = tid & 31;
  const int wv   = tid >> 5;       // 0..15 waves
  const int area = wv >> 3;        // 0: h1 update, 1: h2 update
  const int tq   = wv & 7;         // tile pair index
  const int nl   = lane & 15;
  const int hs   = lane >> 4;
  const int ml   = nl;             // A-fragment row owned by this lane
  const int b0   = blockIdx.x * 16;
  const int gn0  = tq * 32 + nl;   // output columns of the two tiles
  const int gn1  = gn0 + 16;

  // ---- stage W11 / W22 into LDS with the rotation swizzle ----
  for (int i = tid; i < 2 * 256 * 32; i += 512) {
    const int mat = i >> 13;           // 0: W11, 1: W22
    const int rem = i & 8191;
    const int r   = rem >> 5;
    const int k   = (rem & 31) * 8;
    const __bf16* src = (mat ? W22g : W11g) + (size_t)r * Hn + k;
    v8bf v = *(const v8bf*)src;
    *(v8bf*)&sW[mat][r][(k + r * 8) & 255] = v;
  }
  for (int i = tid; i < 2 * 16 * Hn; i += 512)
    (&hbuf[0][0][0][0])[i] = 0;        // zero t=0 read buffer

  // LDS chain:    area0: h1 x W11     area1: h2 x W22
  // global chain: area0: h2 x 0.5W21  area1: h1 x 0.5W12
  const int aL = area;       // hbuf area feeding the LDS-weight chain
  const int aG = area ^ 1;   // hbuf area feeding the global-weight chain
  unsigned long long wg_u =
      uniform_u64((unsigned long long)(area == 0 ? W21 : W12));
  unsigned long long wi_u = (unsigned long long)Wi;  // kernel-arg derived: uniform

  const __bf16* xrow = Xb + (size_t)(b0 + ml) * Tn * In + hs * 16;
  const float bv0 = (area == 0) ? bi[gn0] : 0.0f;
  const float bv1 = (area == 0) ? bi[gn1] : 0.0f;

  float st0[8], st1[8];
#pragma unroll
  for (int v = 0; v < 8; ++v) { st0[v] = 0.0f; st1[v] = 0.0f; }

  __syncthreads();

  for (int t = 0; t < Tn; ++t) {
    const int rb = t & 1, wb = rb ^ 1;

    // defeat LICM: keep weight loads inside the loop (they hit WGP$/L2/LDS);
    // hoisting them spills to scratch at this occupancy.
    int opq = 0;
    asm volatile("" : "+s"(opq));
    asm volatile("" : "+s"(wg_u));
    asm volatile("" : "+s"(wi_u));
    const __bf16* Wg  = (const __bf16*)wg_u;
    const __bf16* Wip = (const __bf16*)wi_u;

    const __bf16* hL = (const __bf16*)&hbuf[rb][aL][ml][0];
    const __bf16* hG = (const __bf16*)&hbuf[rb][aG][ml][0];

    v8f accA0 = {}, accA1 = {};   // LDS-weight chains (+x proj for area0)
    v8f accB0 = {}, accB1 = {};   // global-weight chains

    if (area == 0) {
      const __bf16* xp = xrow + (size_t)t * In;
#pragma unroll
      for (int c = 0; c < 4; ++c) {
        v16bf ax = frag_cat(xp + 32 * c, xp + 32 * c + 8);
        const __bf16* w0 = Wip + (size_t)gn0 * In + hs * 16 + 32 * c;
        v16bf bw0 = frag_cat(w0, w0 + 8);
        accA0 = __builtin_amdgcn_wmma_f32_16x16x32_bf16(false, ax, false, bw0,
                                                        (short)0, accA0, false, false);
        const __bf16* w1 = Wip + (size_t)gn1 * In + hs * 16 + 32 * c;
        v16bf bw1 = frag_cat(w1, w1 + 8);
        accA1 = __builtin_amdgcn_wmma_f32_16x16x32_bf16(false, ax, false, bw1,
                                                        (short)0, accA1, false, false);
      }
    }

#pragma unroll
    for (int c = 0; c < 8; ++c) {
      const int kb = hs * 16 + 32 * c;              // B fragment k base
      const int al = (32 * c + hs * 16 + ml * 8) & 255;       // rotated A lo
      const int ah = (32 * c + 8 + hs * 16 + ml * 8) & 255;  // rotated A hi

      // LDS-weight chain
      v16bf aLf = frag_cat(hL + al, hL + ah);
      const __bf16* r0 = (const __bf16*)&sW[area][gn0][0];
      v16bf bL0 = frag_cat(r0 + ((kb + gn0 * 8 + opq) & 255),
                           r0 + ((kb + 8 + gn0 * 8 + opq) & 255));
      accA0 = __builtin_amdgcn_wmma_f32_16x16x32_bf16(false, aLf, false, bL0,
                                                      (short)0, accA0, false, false);
      const __bf16* r1 = (const __bf16*)&sW[area][gn1][0];
      v16bf bL1 = frag_cat(r1 + ((kb + gn1 * 8 + opq) & 255),
                           r1 + ((kb + 8 + gn1 * 8 + opq) & 255));
      accA1 = __builtin_amdgcn_wmma_f32_16x16x32_bf16(false, aLf, false, bL1,
                                                      (short)0, accA1, false, false);

      // global-weight chain
      v16bf aGf = frag_cat(hG + al, hG + ah);
      const __bf16* g0 = Wg + (size_t)gn0 * Hn + kb;
      v16bf bG0 = frag_cat(g0, g0 + 8);
      accB0 = __builtin_amdgcn_wmma_f32_16x16x32_bf16(false, aGf, false, bG0,
                                                      (short)0, accB0, false, false);
      const __bf16* g1 = Wg + (size_t)gn1 * Hn + kb;
      v16bf bG1 = frag_cat(g1, g1 + 8);
      accB1 = __builtin_amdgcn_wmma_f32_16x16x32_bf16(false, aGf, false, bG1,
                                                      (short)0, accB1, false, false);
    }

    // leaky update + relu; stream outputs, refresh bf16 state
#pragma unroll
    for (int v = 0; v < 8; ++v) {
      const int m = hs * 8 + v;                     // C-fragment row
      const size_t ob = ((size_t)(b0 + m) * Tn + t) * 512 + (size_t)area * 256;

      float p0 = accA0[v] + accB0[v] + bv0;
      float h0 = fmaxf(st0[v] * (1.0f - ADEC) + p0 * ADEC, 0.0f);
      st0[v] = h0;
      __builtin_nontemporal_store(h0, &out[ob + gn0]);
      hbuf[wb][area][m][(gn0 + m * 8) & 255] =
          __builtin_bit_cast(unsigned short, __float2bfloat16(h0));

      float p1 = accA1[v] + accB1[v] + bv1;
      float h1 = fmaxf(st1[v] * (1.0f - ADEC) + p1 * ADEC, 0.0f);
      st1[v] = h1;
      __builtin_nontemporal_store(h1, &out[ob + gn1]);
      hbuf[wb][area][m][(gn1 + m * 8) & 255] =
          __builtin_bit_cast(unsigned short, __float2bfloat16(h1));
    }
    __syncthreads();
  }

  // final h1, h2 states appended after outputs
#pragma unroll
  for (int v = 0; v < 8; ++v) {
    const int m = hs * 8 + v;
    const size_t base = (size_t)Bn * Tn * 512 + (size_t)area * Bn * Hn +
                        (size_t)(b0 + m) * Hn;
    out[base + gn0] = st0[v];
    out[base + gn1] = st1[v];
  }
}

extern "C" void kernel_launch(void* const* d_in, const int* in_sizes, int n_in,
                              void* d_out, int out_size, void* d_ws, size_t ws_size,
                              hipStream_t stream) {
  (void)in_sizes; (void)n_in; (void)out_size; (void)ws_size;
  const float* x   = (const float*)d_in[0];
  // d_in[1] = seq_lengths: forward no-op (stop_gradient identity), unused.
  const float* Wi  = (const float*)d_in[2];
  const float* bi  = (const float*)d_in[3];
  const float* W11 = (const float*)d_in[4];
  const float* W22 = (const float*)d_in[5];
  const float* W12 = (const float*)d_in[6];
  const float* W21 = (const float*)d_in[7];

  unsigned short* ws = (unsigned short*)d_ws;
  float* out = (float*)d_out;

  auto cvt = [&](const float* src, size_t dst_off, int n, float scale) {
    cvt_bf16_kernel<<<(n + 255) / 256, 256, 0, stream>>>(src, ws + dst_off, n, scale);
  };
  cvt(Wi,  OFF_WI,  Hn * In, 1.0f);
  cvt(W11, OFF_W11, Hn * Hn, 1.0f);
  cvt(W22, OFF_W22, Hn * Hn, 1.0f);
  cvt(W12, OFF_W12, Hn * Hn, 0.5f);   // fold ALPHA_FF
  cvt(W21, OFF_W21, Hn * Hn, 0.5f);   // fold ALPHA_FB
  cvt(x,   OFF_X,   Bn * Tn * In, 1.0f);

  ctrnn_scan_kernel<<<4, 512, 0, stream>>>(ws, bi, out);
}
